// PathAttention_86930138071271
// MI455X (gfx1250) — matrix operand order repaired and checked by reading, not compiled, not run on hardware
//
#include <hip/hip_runtime.h>
#include <hip/hip_bf16.h>
#include <math.h>

typedef __attribute__((ext_vector_type(16))) _Float16 v16h;
typedef __attribute__((ext_vector_type(8)))  _Float16 v8h;
typedef __attribute__((ext_vector_type(8)))  float    v8f;

#define BATCH 2048
#define ITER  4                   // batches per persistent block
#define NBLK  (BATCH / ITER)      // 512 blocks
#define LSEQ  512
#define D2    64                  // 2*embed
#define D4    128                 // 4*embed
#define NHEAD 8
#define NEG_BIG (-1.0e9f)

// LDS row strides (in halves); all multiples of 8 halves (16 B) so v8h
// fragment loads stay 16B-aligned and lower to ds_load_b128.
#define VSTR  72                  // sValH row stride (64 + 8 pad)
#define W1STR 72
#define XSTR  136                 // 128 + 8 pad
#define W2STR 136
#define TSTR  520                 // sValT row stride (512 + 8 pad)
#define PSTR  520                 // sPT   row stride

static __device__ inline v16h mk16(v8h lo, v8h hi) {
  v16h r;
#pragma unroll
  for (int i = 0; i < 8; ++i) { r[i] = lo[i]; r[i + 8] = hi[i]; }
  return r;
}

__global__ __launch_bounds__(256) void path_attn_kernel(
    const float* __restrict__ value,   // [B, L, 64]
    const float* __restrict__ mask,    // [B, L]
    const float* __restrict__ w1,      // [128, 64]
    const float* __restrict__ w2,      // [8, 128]
    float* __restrict__ out)           // [B, 8, 64]
{
  __shared__ __align__(16) _Float16 sValH[LSEQ * VSTR];    // ~72 KB  value (A for GEMM1)
  __shared__ __align__(16) _Float16 sValT[D2 * TSTR];      // ~65 KB  value^T (B for GEMM3)
  __shared__ __align__(16) _Float16 sW1[D4 * W1STR];       // 18 KB
  __shared__ __align__(16) _Float16 sW2[16 * W2STR];       // ~4.3 KB (heads padded to 16)
  __shared__ __align__(16) _Float16 sX[8][16 * XSTR];      // ~34 KB  per-wave relu(x) staging
  __shared__ __align__(16) _Float16 sPT[16 * PSTR];        // ~16 KB  exp(s-max)^T (A for GEMM3)
  __shared__ __align__(16) float    sSc[LSEQ * 17];        // ~34 KB  raw scores
  __shared__ __align__(16) float    sPart[2 * 16 * 68];    // ~8.7 KB GEMM3 K-split partials
  __shared__ __align__(16) float    sMask[LSEQ];           // 2 KB   (pre-multiplied by -1e9)
  __shared__ float                  sInv[NHEAD];           // 1/Z per head

  const int tid  = threadIdx.x;
  const int lane = tid & 31;
  const int wave = tid >> 5;
  const int mrow = lane & 15;   // row (A) or column (B/C) index within a tile
  const int hi   = lane >> 4;   // lane-half selector per WMMA VGPR layout

  // ---- One-time staging of loop-invariant weights ----
  for (int i = tid; i < 16 * W2STR; i += 256)
    sW2[i] = (_Float16)0.0f;                          // pad heads 8..15 with zeros
  for (int i = tid; i < 8 * PSTR; i += 256)
    sPT[8 * PSTR + i] = (_Float16)0.0f;               // zero P rows 8..15 once
  __syncthreads();                                    // zero-fill before scatter
  for (int i = tid; i < D4 * D2; i += 256)
    sW1[(i / D2) * W1STR + (i % D2)] = (_Float16)w1[i];
  for (int i = tid; i < NHEAD * D4; i += 256)
    sW2[(i / D4) * W2STR + (i % D4)] = (_Float16)w2[i];

  for (int it = 0; it < ITER; ++it) {
    const int b = blockIdx.x + it * NBLK;             // this block's batch

    // ---------------- Phase 0: stage value/mask into LDS ----------------
    {
      const float4* vb4 = (const float4*)(value + (size_t)b * LSEQ * D2);
      for (int i = tid; i < (LSEQ * D2) / 4; i += 256) {
        float4 v = vb4[i];
        int l = (i * 4) / D2, d = (i * 4) % D2;
        _Float16 h0 = (_Float16)v.x, h1 = (_Float16)v.y;
        _Float16 h2 = (_Float16)v.z, h3 = (_Float16)v.w;
        _Float16* p = &sValH[l * VSTR + d];
        p[0] = h0; p[1] = h1; p[2] = h2; p[3] = h3;
        sValT[(d + 0) * TSTR + l] = h0;
        sValT[(d + 1) * TSTR + l] = h1;
        sValT[(d + 2) * TSTR + l] = h2;
        sValT[(d + 3) * TSTR + l] = h3;
      }
      for (int i = tid; i < LSEQ; i += 256)
        sMask[i] = NEG_BIG * mask[(size_t)b * LSEQ + i];
    }
    __syncthreads();

    // ---- L2 prefetch of next iteration's value tile (global_prefetch_b8) ----
    if (it + 1 < ITER) {
      const char* np = (const char*)(value + (size_t)(b + NBLK) * LSEQ * D2);
#pragma unroll
      for (int j = 0; j < 4; ++j)                     // 1024 cachelines of 128 B
        __builtin_prefetch(np + (tid + j * 256) * 128, 0, 3);
    }

    // ------- Phase 1: x = relu(value @ w1^T); scores = x @ w2^T (per l-tile) -------
    for (int mt = wave; mt < LSEQ / 16; mt += 8) {
#pragma unroll
      for (int nt = 0; nt < D4 / 16; ++nt) {
        v8f c = {};
#pragma unroll
        for (int kk = 0; kk < 2; ++kk) {
          const _Float16* ap = &sValH[(mt * 16 + mrow) * VSTR + kk * 32 + hi * 8];
          v16h a = mk16(*(const v8h*)ap, *(const v8h*)(ap + 16));
          const _Float16* bp = &sW1[(nt * 16 + mrow) * W1STR + kk * 32 + hi * 8];
          v16h bb = mk16(*(const v8h*)bp, *(const v8h*)(bp + 16));
          c = __builtin_amdgcn_wmma_f32_16x16x32_f16(false, a, false, bb,
                                                     (short)0, c, false, false);
        }
#pragma unroll
        for (int r = 0; r < 8; ++r) {
          float x = c[r] > 0.0f ? c[r] : 0.0f;
          sX[wave][(r + 8 * hi) * XSTR + nt * 16 + mrow] = (_Float16)x;
        }
      }
      v8f sc = {};
#pragma unroll
      for (int kk = 0; kk < 4; ++kk) {
        const _Float16* ap = &sX[wave][mrow * XSTR + kk * 32 + hi * 8];
        v16h a = mk16(*(const v8h*)ap, *(const v8h*)(ap + 16));
        const _Float16* bp = &sW2[mrow * W2STR + kk * 32 + hi * 8];
        v16h bb = mk16(*(const v8h*)bp, *(const v8h*)(bp + 16));
        sc = __builtin_amdgcn_wmma_f32_16x16x32_f16(false, a, false, bb,
                                                    (short)0, sc, false, false);
      }
#pragma unroll
      for (int r = 0; r < 8; ++r) {
        int l = mt * 16 + r + 8 * hi;
        sSc[l * 17 + mrow] = sc[r] + sMask[l];        // lane column = head
      }
    }
    __syncthreads();

    // ------- Phase 2: softmax stats over L, one wave per head; emit P^T f16 -------
    {
      const int h = wave;                             // 8 waves == 8 heads
      float vloc[16];
      float m = -INFINITY;
#pragma unroll
      for (int j = 0; j < 16; ++j) {
        float s = sSc[(lane + 32 * j) * 17 + h];
        vloc[j] = s;
        m = fmaxf(m, s);
      }
#pragma unroll
      for (int off = 16; off; off >>= 1) m = fmaxf(m, __shfl_xor(m, off, 32));
      float z = 0.0f;
#pragma unroll
      for (int j = 0; j < 16; ++j) {
        float e = __expf(vloc[j] - m);
        sPT[h * PSTR + lane + 32 * j] = (_Float16)e;  // unnormalized prob, f16-safe (0,1]
        z += e;
      }
#pragma unroll
      for (int off = 16; off; off >>= 1) z += __shfl_xor(z, off, 32);
      if (lane == 0) sInv[h] = 1.0f / z;
    }
    __syncthreads();

    // ------- Phase 3: out = P @ value via WMMA (M=16pad, N=64, K=512, K-split x2) ----
    {
      const int nt    = wave & 3;                     // n-tile: d = nt*16 .. +15
      const int khalf = wave >> 2;                    // K half: l = khalf*256 .. +255
      v8f c = {};
#pragma unroll
      for (int kk = 0; kk < 8; ++kk) {
        const int kb = khalf * 256 + kk * 32;
        const _Float16* ap = &sPT[mrow * PSTR + kb + hi * 8];
        v16h a = mk16(*(const v8h*)ap, *(const v8h*)(ap + 16));
        const _Float16* bp = &sValT[(nt * 16 + mrow) * TSTR + kb + hi * 8];
        v16h bb = mk16(*(const v8h*)bp, *(const v8h*)(bp + 16));
        c = __builtin_amdgcn_wmma_f32_16x16x32_f16(false, a, false, bb,
                                                   (short)0, c, false, false);
      }
#pragma unroll
      for (int r = 0; r < 8; ++r)                     // C row = head, col = d-in-tile
        sPart[(khalf * 16 + r + 8 * hi) * 68 + nt * 16 + mrow] = c[r];
    }
    __syncthreads();

    // ------- Combine K-split partials, scale by 1/Z, write out -------
#pragma unroll
    for (int rep = 0; rep < 2; ++rep) {
      int idx = tid + rep * 256;                      // 512 outputs, 2 per thread
      int h = idx >> 6, d = idx & 63;
      float r = (sPart[(0 * 16 + h) * 68 + d] + sPart[(1 * 16 + h) * 68 + d]) * sInv[h];
      out[(size_t)b * (NHEAD * D2) + idx] = r;
    }
    __syncthreads();                                  // protect LDS before next batch
  }
}

extern "C" void kernel_launch(void* const* d_in, const int* in_sizes, int n_in,
                              void* d_out, int out_size, void* d_ws, size_t ws_size,
                              hipStream_t stream) {
  (void)in_sizes; (void)n_in; (void)d_ws; (void)ws_size; (void)out_size;
  const float* value = (const float*)d_in[0];
  const float* mask  = (const float*)d_in[1];
  const float* w1    = (const float*)d_in[2];
  const float* w2    = (const float*)d_in[3];
  float* out = (float*)d_out;
  hipLaunchKernelGGL(path_attn_kernel, dim3(NBLK), dim3(256), 0, stream,
                     value, mask, w1, w2, out);
}
